// SED_Attention_LSTM_50027779064391
// MI455X (gfx1250) — matrix operand order repaired and checked by reading, not compiled, not run on hardware
//
#include <hip/hip_runtime.h>
#include <hip/hip_bf16.h>

typedef _Float16 half8  __attribute__((ext_vector_type(8)));
typedef _Float16 half4h __attribute__((ext_vector_type(4)));
typedef _Float16 v16h   __attribute__((ext_vector_type(16)));
typedef float    v8f    __attribute__((ext_vector_type(8)));

__device__ __forceinline__ v8f vzero8() {
    v8f z = {0.f,0.f,0.f,0.f,0.f,0.f,0.f,0.f};
    return z;
}

// ---- WMMA fragment loaders (layouts per CDNA5 ISA 7.12.2) ----
// A-matrix 16x32 f16 (M x K), S row-major with stride 128 halves.
// lanes 0-15: M=lane, elems 0-7 = K 0..7,  elems 8-15 = K 16..23
// lanes16-31: M=lane-16, elems 0-7 = K 8..15, elems 8-15 = K 24..31
__device__ __forceinline__ v16h frag_a(const _Float16* S, int lane, int kt) {
    int row = lane & 15;
    int hi  = (lane >> 4) & 1;
    int k0  = kt * 32 + hi * 8;
    const _Float16* p = S + row * 128 + k0;
    half8 lo = *(const half8*)(p);
    half8 hh = *(const half8*)(p + 16);
    return __builtin_shufflevector(lo, hh, 0,1,2,3,4,5,6,7,8,9,10,11,12,13,14,15);
}

// B-matrix 32x16 f16 (K x N) built from W (N x K row-major, stride 128):
// B[k][n] = W[n][k].  lanes 0-15: N=lane, elems 0-7 = K 0..7, elems 8-15 = K 16..23;
// lanes 16-31: N=lane-16, elems 0-7 = K 8..15, elems 8-15 = K 24..31.
__device__ __forceinline__ v16h frag_b(const _Float16* W, int lane, int nt, int kt) {
    int n  = nt * 16 + (lane & 15);
    int hi = (lane >> 4) & 1;
    const _Float16* p = W + n * 128 + kt * 32 + hi * 8;
    half8 lo = *(const half8*)(p);
    half8 hh = *(const half8*)(p + 16);
    return __builtin_shufflevector(lo, hh, 0,1,2,3,4,5,6,7,8,9,10,11,12,13,14,15);
}

__device__ __forceinline__ v8f wmma16(v16h a, v16h b, v8f c) {
    return __builtin_amdgcn_wmma_f32_16x16x32_f16(false, a, false, b, (short)0, c, false, false);
}

// fast transcendentals: v_exp_f32 + v_rcp_f32, no div_scale chains
__device__ __forceinline__ float sigmf(float x) {
    return __builtin_amdgcn_rcpf(1.f + __expf(-x));
}
__device__ __forceinline__ float tanhsf(float x) {
    float e = __expf(-2.f * fabsf(x));
    float t = (1.f - e) * __builtin_amdgcn_rcpf(1.f + e);
    return x >= 0.f ? t : -t;
}

// ---------------- small prep kernels ----------------
__global__ void k_f32_to_f16(const float* __restrict__ s, _Float16* __restrict__ d, int n) {
    int i = blockIdx.x * 256 + threadIdx.x;
    if (i < n) d[i] = (_Float16)s[i];
}

// pack W_out (20x128) into padded 32x128 f16
__global__ void k_wout_pack(const float* __restrict__ s, _Float16* __restrict__ d) {
    int i = blockIdx.x * 256 + threadIdx.x;
    if (i < 32 * 128) d[i] = (i < 20 * 128) ? (_Float16)s[i] : (_Float16)0.f;
}

// alpha = sum W_v[k]*W_ao[k]; beta = sum b_v[k]*W_ao[k] + b_ao
__global__ void k_alphabeta(const float* __restrict__ Wv, const float* __restrict__ bv,
                            const float* __restrict__ Wao, const float* __restrict__ bao,
                            float* __restrict__ ab) {
    if (threadIdx.x == 0 && blockIdx.x == 0) {
        float a = 0.f, b = 0.f;
        for (int k = 0; k < 128; ++k) { a += Wv[k] * Wao[k]; b += bv[k] * Wao[k]; }
        ab[0] = a; ab[1] = b + bao[0];
    }
}

// input projection: XP[(t*8+b)][i] = sum_mel x[b][mel][t]*W_in[i][mel] + b_in[i]  (f16 out)
__global__ void k_inproj(const float* __restrict__ x, const float* __restrict__ Win,
                         const float* __restrict__ bin, _Float16* __restrict__ XP) {
    int idx = blockIdx.x * 256 + threadIdx.x;       // 4096*128 outputs
    int row = idx >> 7, i = idx & 127;
    int t = row >> 3, b = row & 7;
    float acc = bin[i];
    const float* xb = x + (size_t)b * 64 * 512 + t;
    const float* wr = Win + i * 64;
#pragma unroll 8
    for (int mel = 0; mel < 64; ++mel) acc += xb[(size_t)mel * 512] * wr[mel];
    XP[idx] = (_Float16)acc;
}

// ---------------- generic M=4096, N=128, K=128 WMMA GEMM:  C = A @ W^T + bias (+bias2) ----
// MODE 0: f32 out in scan-friendly [t][n][b] layout, bias + bias2
// MODE 1: f32 out, row-major [m][n], single bias
// MODE 2: f16 sigmoid out, row-major [m][n], single bias
template <int MODE>
__global__ __launch_bounds__(256)
void k_gemm128(const _Float16* __restrict__ A, const _Float16* __restrict__ W,
               const float* __restrict__ bias, const float* __restrict__ bias2,
               float* __restrict__ of32, _Float16* __restrict__ of16) {
    int tid = threadIdx.x, lane = tid & 31, wv = tid >> 5;
    int mt = blockIdx.x;
    int col = wv * 16 + (lane & 15);
    int hi  = (lane >> 4) & 1;
    v8f acc = vzero8();
    const _Float16* At = A + (size_t)mt * 16 * 128;
#pragma unroll
    for (int kt = 0; kt < 4; ++kt)
        acc = wmma16(frag_a(At, lane, kt), frag_b(W, lane, wv, kt), acc);
    float bb = bias[col];
    if (MODE == 0) bb += bias2[col];
#pragma unroll
    for (int r = 0; r < 8; ++r) {
        int m = mt * 16 + hi * 8 + r;
        float v = acc[r] + bb;
        if (MODE == 0) {
            int t = m >> 3, b = m & 7;                      // [t][n][b] layout
            of32[((size_t)t * 128 + col) * 8 + b] = v;
        } else if (MODE == 1) {
            of32[(size_t)m * 128 + col] = v;
        } else {
            of16[(size_t)m * 128 + col] = (_Float16)sigmf(v);
        }
    }
}

// ---------------- output GEMM: Y[b][c][t] = HIST @ W_out^T + b_out  (N padded to 32) ----------------
__global__ __launch_bounds__(64)
void k_gemm_out(const _Float16* __restrict__ HIST, const _Float16* __restrict__ Wout,
                const float* __restrict__ bout, float* __restrict__ OUT) {
    int tid = threadIdx.x, lane = tid & 31, wv = tid >> 5;   // 2 waves -> N=32
    int mt = blockIdx.x;
    int c  = wv * 16 + (lane & 15);
    int hi = (lane >> 4) & 1;
    v8f acc = vzero8();
    const _Float16* At = HIST + (size_t)mt * 16 * 128;
#pragma unroll
    for (int kt = 0; kt < 4; ++kt)
        acc = wmma16(frag_a(At, lane, kt), frag_b(Wout, lane, wv, kt), acc);
    if (c < 20) {
        float bb = bout[c];
#pragma unroll
        for (int r = 0; r < 8; ++r) {
            int m = mt * 16 + hi * 8 + r;
            int t = m >> 3, b = m & 7;
            OUT[((size_t)b * 20 + c) * 512 + t] = acc[r] + bb;
        }
    }
}

// ---------------- persistent recurrent scan: 1 block, 8 waves, weights in LDS ----------------
__global__ __launch_bounds__(256)
void k_scan(const _Float16* __restrict__ Wh16 /* Whi,Whf,Whg,Who,Wk contiguous */,
            const float* __restrict__ Xi, const float* __restrict__ Xf,
            const float* __restrict__ Xg, const float* __restrict__ Xo,
            const float* __restrict__ Q,  const float* __restrict__ bk,
            const float* __restrict__ ab, _Float16* __restrict__ HIST) {
    extern __shared__ char smem[];
    _Float16* WhL = (_Float16*)smem;           // 4*16384 halfs  (LDS byte offset 0)
    _Float16* WkL = WhL + 4 * 16384;           // 16384 halfs
    _Float16* hA  = WkL + 16384;               // 16x128 (rows 8..15 stay 0)
    _Float16* cA  = hA + 2048;                 // 16x128
    float* koF  = (float*)(cA + 2048);         // 8x128 x {k_scaled, o} interleaved
    float* pmin = koF + 2048;                  // 64
    float* pmax = pmin + 64;                   // 64

    int tid = threadIdx.x, lane = tid & 31, wv = tid >> 5;
    int col = wv * 16 + (lane & 15);

    // stage 160KB of recurrent weights straight into LDS via the CDNA5 async
    // global->LDS path (no VGPR roundtrip, tracked by ASYNCcnt).
    // Dynamic LDS base is offset 0 (no static LDS in this kernel).
    for (int i = tid; i < 10240; i += 256) {
        unsigned    dst = (unsigned)i * 16u;                  // LDS byte offset
        const void* src = (const char*)Wh16 + (size_t)i * 16; // global address
        asm volatile("global_load_async_to_lds_b128 %0, %1, off"
                     :: "v"(dst), "v"(src) : "memory");
    }
    // zero hA + cA (8192 bytes)
    for (int i = tid; i < 512; i += 256)
        ((float4*)hA)[i] = make_float4(0.f, 0.f, 0.f, 0.f);
    asm volatile("s_wait_asynccnt 0x0" ::: "memory");
    __syncthreads();

    float creg[8];
#pragma unroll
    for (int r = 0; r < 8; ++r) creg[r] = 0.f;
    float bkv   = bk[col];
    float alpha = ab[0], beta = ab[1];
    const float inv_scale = 0.0883883476483184f;   // 1/sqrt(128)

    const _Float16* Wi = WhL;
    const _Float16* Wf = WhL + 16384;
    const _Float16* Wg = WhL + 32768;
    const _Float16* Wo = WhL + 49152;

    // softmax-phase mapping: wave = batch row, 4 consecutive i per lane
    int sb  = tid >> 5;           // batch 0..7
    int si0 = (tid & 31) * 4;     // 4 consecutive columns

    for (int t = 0; t < 512; ++t) {
        size_t base = (size_t)t * 1024;            // per-t block of 8*128 floats
        // speculative prefetch of next timestep's precomputed rows
        __builtin_prefetch(Xi + base + 1024 + col * 8, 0, 1);
        __builtin_prefetch(Xf + base + 1024 + col * 8, 0, 1);
        __builtin_prefetch(Xg + base + 1024 + col * 8, 0, 1);
        __builtin_prefetch(Xo + base + 1024 + col * 8, 0, 1);
        __builtin_prefetch(Q  + base + 1024 + (size_t)sb * 128 + si0, 0, 1);

        // x-parts in [t][n][b] layout: 8 contiguous floats per (t,n) -> 2x b128 each
        float4 xi0 = *(const float4*)(Xi + base + (size_t)col * 8);
        float4 xi1 = *(const float4*)(Xi + base + (size_t)col * 8 + 4);
        float4 xf0 = *(const float4*)(Xf + base + (size_t)col * 8);
        float4 xf1 = *(const float4*)(Xf + base + (size_t)col * 8 + 4);
        float4 xg0 = *(const float4*)(Xg + base + (size_t)col * 8);
        float4 xg1 = *(const float4*)(Xg + base + (size_t)col * 8 + 4);
        float4 xo0 = *(const float4*)(Xo + base + (size_t)col * 8);
        float4 xo1 = *(const float4*)(Xo + base + (size_t)col * 8 + 4);
        float xi[8] = {xi0.x,xi0.y,xi0.z,xi0.w, xi1.x,xi1.y,xi1.z,xi1.w};
        float xf[8] = {xf0.x,xf0.y,xf0.z,xf0.w, xf1.x,xf1.y,xf1.z,xf1.w};
        float xg[8] = {xg0.x,xg0.y,xg0.z,xg0.w, xg1.x,xg1.y,xg1.z,xg1.w};
        float xo[8] = {xo0.x,xo0.y,xo0.z,xo0.w, xo1.x,xo1.y,xo1.z,xo1.w};

        // gates: h @ W^T  (16 WMMAs per wave)
        v8f ai = vzero8(), af = vzero8(), ag = vzero8(), ao = vzero8();
#pragma unroll
        for (int kt = 0; kt < 4; ++kt) {
            v16h a = frag_a(hA, lane, kt);
            ai = wmma16(a, frag_b(Wi, lane, wv, kt), ai);
            af = wmma16(a, frag_b(Wf, lane, wv, kt), af);
            ag = wmma16(a, frag_b(Wg, lane, wv, kt), ag);
            ao = wmma16(a, frag_b(Wo, lane, wv, kt), ao);
        }
#pragma unroll
        for (int r = 0; r < 8; ++r) {
            float iv = sigmf(ai[r] + xi[r]);
            float fv = sigmf(af[r] + xf[r]);
            float gv = tanhsf(ag[r] + xg[r]);
            float ov = sigmf(ao[r] + xo[r]);
            float cn = fv * creg[r] + iv * gv;
            creg[r] = cn;
            if (lane < 16) {                       // valid batch rows only
                cA[r * 128 + col] = (_Float16)cn;
                koF[(r * 128 + col) * 2 + 1] = ov;
            }
        }
        __syncthreads();

        // k = c_new @ W_k^T  (4 WMMAs per wave), pre-scale by 1/sqrt(H)
        v8f ak = vzero8();
#pragma unroll
        for (int kt = 0; kt < 4; ++kt)
            ak = wmma16(frag_a(cA, lane, kt), frag_b(WkL, lane, wv, kt), ak);
        if (lane < 16) {
#pragma unroll
            for (int r = 0; r < 8; ++r)
                koF[(r * 128 + col) * 2] = (ak[r] + bkv) * inv_scale;
        }
        __syncthreads();

        // per-batch partial min/max of scaled k (softmax stability), 8 partials per batch
        if (tid < 64) {
            int b = tid >> 3, j0 = (tid & 7) * 16;
            float mn = 3.4e38f, mx = -3.4e38f;
#pragma unroll
            for (int j = 0; j < 16; ++j) {
                float v = koF[(b * 128 + j0 + j) * 2];
                mn = fminf(mn, v); mx = fmaxf(mx, v);
            }
            pmin[tid] = mn; pmax[tid] = mx;
        }
        __syncthreads();

        // softmax-weighted sum of o, collapsed attention: h_new = alpha*s + beta
        // each lane: 4 consecutive i of ONE batch row -> k/o LDS loads shared
        {
            float kmn = 3.4e38f, kmx = -3.4e38f;
#pragma unroll
            for (int j = 0; j < 8; ++j) {
                kmn = fminf(kmn, pmin[sb * 8 + j]);
                kmx = fmaxf(kmx, pmax[sb * 8 + j]);
            }
            float4 qv = *(const float4*)(Q + base + (size_t)sb * 128 + si0);
            float m0 = (qv.x >= 0.f) ? qv.x * kmx : qv.x * kmn;
            float m1 = (qv.y >= 0.f) ? qv.y * kmx : qv.y * kmn;
            float m2 = (qv.z >= 0.f) ? qv.z * kmx : qv.z * kmn;
            float m3 = (qv.w >= 0.f) ? qv.w * kmx : qv.w * kmn;
            float n0 = 0.f, n1 = 0.f, n2 = 0.f, n3 = 0.f;
            float d0 = 0.f, d1 = 0.f, d2 = 0.f, d3 = 0.f;
            const float2* ko = (const float2*)koF + sb * 128;
#pragma unroll 4
            for (int j = 0; j < 128; ++j) {
                float2 kv = ko[j];                 // single b64 broadcast load
                float e0 = __expf(qv.x * kv.x - m0);
                float e1 = __expf(qv.y * kv.x - m1);
                float e2 = __expf(qv.z * kv.x - m2);
                float e3 = __expf(qv.w * kv.x - m3);
                n0 += e0 * kv.y; d0 += e0;
                n1 += e1 * kv.y; d1 += e1;
                n2 += e2 * kv.y; d2 += e2;
                n3 += e3 * kv.y; d3 += e3;
            }
            half4h hn;
            hn.x = (_Float16)(alpha * (n0 * __builtin_amdgcn_rcpf(d0)) + beta);
            hn.y = (_Float16)(alpha * (n1 * __builtin_amdgcn_rcpf(d1)) + beta);
            hn.z = (_Float16)(alpha * (n2 * __builtin_amdgcn_rcpf(d2)) + beta);
            hn.w = (_Float16)(alpha * (n3 * __builtin_amdgcn_rcpf(d3)) + beta);
            *(half4h*)(hA + sb * 128 + si0) = hn;                 // 8B ds store
            *(half4h*)(HIST + base + (size_t)sb * 128 + si0) = hn; // 8B global store
        }
        __syncthreads();
    }
}

// ---------------- workspace layout (bytes) ----------------
#define WS_WHI   0u
#define WS_WHF   32768u
#define WS_WHG   65536u
#define WS_WHO   98304u
#define WS_WK    131072u            // Whi..Wk contiguous: 163840 B
#define WS_WII   163840u
#define WS_WIF   196608u
#define WS_WIG   229376u
#define WS_WIO   262144u
#define WS_WXO   294912u
#define WS_WQ    327680u
#define WS_WOUT  360448u            // 32x128 f16 = 8192
#define WS_AB    368640u            // 2 floats
#define WS_XP16  368896u            // 4096*128 f16
#define WS_XO16  1417472u
#define WS_XI32  2466048u           // 4096*128 f32  ([t][n][b] layout)
#define WS_XF32  4563200u
#define WS_XG32  6660352u
#define WS_XOG32 8757504u
#define WS_Q32   10854656u
#define WS_HIST  12951808u          // 4096*128 f16 -> end 14000384

#define SCAN_LDS 180736u

extern "C" void kernel_launch(void* const* d_in, const int* in_sizes, int n_in,
                              void* d_out, int out_size, void* d_ws, size_t ws_size,
                              hipStream_t stream) {
    (void)in_sizes; (void)n_in; (void)out_size; (void)ws_size;
    const float* x     = (const float*)d_in[0];
    const float* W_in  = (const float*)d_in[1];  const float* b_in = (const float*)d_in[2];
    const float* W_out = (const float*)d_in[3];  const float* b_out= (const float*)d_in[4];
    const float* W_ii  = (const float*)d_in[5];  const float* b_ii = (const float*)d_in[6];
    const float* W_hi  = (const float*)d_in[7];  const float* b_hi = (const float*)d_in[8];
    const float* W_if  = (const float*)d_in[9];  const float* b_if = (const float*)d_in[10];
    const float* W_hf  = (const float*)d_in[11]; const float* b_hf = (const float*)d_in[12];
    const float* W_ig  = (const float*)d_in[13]; const float* b_ig = (const float*)d_in[14];
    const float* W_hg  = (const float*)d_in[15]; const float* b_hg = (const float*)d_in[16];
    const float* W_io  = (const float*)d_in[17]; const float* b_io = (const float*)d_in[18];
    const float* W_ho  = (const float*)d_in[19]; const float* b_ho = (const float*)d_in[20];
    const float* W_xo  = (const float*)d_in[21]; const float* b_xo = (const float*)d_in[22];
    const float* W_q   = (const float*)d_in[23]; const float* b_q  = (const float*)d_in[24];
    const float* W_k   = (const float*)d_in[25]; const float* b_k  = (const float*)d_in[26];
    const float* W_v   = (const float*)d_in[27]; const float* b_v  = (const float*)d_in[28];
    const float* W_ao  = (const float*)d_in[29]; const float* b_ao = (const float*)d_in[30];

    char* ws = (char*)d_ws;
    _Float16* WHI16  = (_Float16*)(ws + WS_WHI);
    _Float16* WHF16  = (_Float16*)(ws + WS_WHF);
    _Float16* WHG16  = (_Float16*)(ws + WS_WHG);
    _Float16* WHO16  = (_Float16*)(ws + WS_WHO);
    _Float16* WK16   = (_Float16*)(ws + WS_WK);
    _Float16* WII16  = (_Float16*)(ws + WS_WII);
    _Float16* WIF16  = (_Float16*)(ws + WS_WIF);
    _Float16* WIG16  = (_Float16*)(ws + WS_WIG);
    _Float16* WIO16  = (_Float16*)(ws + WS_WIO);
    _Float16* WXO16  = (_Float16*)(ws + WS_WXO);
    _Float16* WQ16   = (_Float16*)(ws + WS_WQ);
    _Float16* WOUT16 = (_Float16*)(ws + WS_WOUT);
    float*    AB     = (float*)   (ws + WS_AB);
    _Float16* XP16   = (_Float16*)(ws + WS_XP16);
    _Float16* XO16   = (_Float16*)(ws + WS_XO16);
    float*    XI32   = (float*)   (ws + WS_XI32);
    float*    XF32   = (float*)   (ws + WS_XF32);
    float*    XG32   = (float*)   (ws + WS_XG32);
    float*    XOG32  = (float*)   (ws + WS_XOG32);
    float*    Q32    = (float*)   (ws + WS_Q32);
    _Float16* HIST16 = (_Float16*)(ws + WS_HIST);

    // weight conversions (f32 -> f16)
    k_f32_to_f16<<<64, 256, 0, stream>>>(W_hi, WHI16, 16384);
    k_f32_to_f16<<<64, 256, 0, stream>>>(W_hf, WHF16, 16384);
    k_f32_to_f16<<<64, 256, 0, stream>>>(W_hg, WHG16, 16384);
    k_f32_to_f16<<<64, 256, 0, stream>>>(W_ho, WHO16, 16384);
    k_f32_to_f16<<<64, 256, 0, stream>>>(W_k,  WK16,  16384);
    k_f32_to_f16<<<64, 256, 0, stream>>>(W_ii, WII16, 16384);
    k_f32_to_f16<<<64, 256, 0, stream>>>(W_if, WIF16, 16384);
    k_f32_to_f16<<<64, 256, 0, stream>>>(W_ig, WIG16, 16384);
    k_f32_to_f16<<<64, 256, 0, stream>>>(W_io, WIO16, 16384);
    k_f32_to_f16<<<64, 256, 0, stream>>>(W_xo, WXO16, 16384);
    k_f32_to_f16<<<64, 256, 0, stream>>>(W_q,  WQ16,  16384);
    k_wout_pack <<<16, 256, 0, stream>>>(W_out, WOUT16);
    k_alphabeta <<<1, 32, 0, stream>>>(W_v, b_v, W_ao, b_ao, AB);

    // parallel precompute of everything x-dependent
    k_inproj<<<2048, 256, 0, stream>>>(x, W_in, b_in, XP16);
    k_gemm128<0><<<256, 256, 0, stream>>>(XP16, WII16, b_ii, b_hi, XI32,  nullptr);
    k_gemm128<0><<<256, 256, 0, stream>>>(XP16, WIF16, b_if, b_hf, XF32,  nullptr);
    k_gemm128<0><<<256, 256, 0, stream>>>(XP16, WIG16, b_ig, b_hg, XG32,  nullptr);
    k_gemm128<0><<<256, 256, 0, stream>>>(XP16, WIO16, b_io, b_ho, XOG32, nullptr);
    k_gemm128<2><<<256, 256, 0, stream>>>(XP16, WXO16, b_xo, nullptr, nullptr, XO16);
    k_gemm128<1><<<256, 256, 0, stream>>>(XO16, WQ16,  b_q,  nullptr, Q32,   nullptr);

    // persistent sequential scan (1 workgroup, LDS-resident weights)
    (void)hipFuncSetAttribute((const void*)k_scan,
                              hipFuncAttributeMaxDynamicSharedMemorySize, (int)SCAN_LDS);
    k_scan<<<1, 256, SCAN_LDS, stream>>>(WHI16, XI32, XF32, XG32, XOG32, Q32, b_k, AB, HIST16);

    // output projection + transpose to (B, C, T)
    k_gemm_out<<<256, 64, 0, stream>>>(HIST16, WOUT16, b_out, (float*)d_out);
}
